// ArtistAttentionProcessor_28200755266144
// MI455X (gfx1250) — compile-verified, matrix operand order checked by reading, not run on hardware
//
#include <hip/hip_runtime.h>
#include <hip/hip_bf16.h>

typedef __attribute__((ext_vector_type(16))) __bf16 v16bf;
typedef __attribute__((ext_vector_type(4)))  __bf16 bf4v;
typedef __attribute__((ext_vector_type(2)))  __bf16 bf2;
typedef __attribute__((ext_vector_type(8)))  float  v8f;

#define BB   6
#define SS   2048
#define CC   320
#define HH   8
#define DH   40      // head dim
#define KROW 72      // K-tile LDS row stride (bf16): 144B rows, every 16B chunk aligned
#define VROW 48      // V-tile LDS row stride (bf16): 96B rows

// K index base for bf16 WMMA A/B operand VGPR v (pairs), per ISA 7.12.2:
// lanes 0-15: VGPR0..3 -> K=0,2,4,6 ; VGPR4..7 -> K=16,18,20,22 ; lanes 16-31: +8
__device__ __forceinline__ int kb_of(int v, int half) {
    return (v < 4 ? 2 * v : 8 + 2 * v) + 8 * half;
}

#define WMMA_BF16(A, Bf, Cf) \
    __builtin_amdgcn_wmma_f32_16x16x32_bf16(false, (A), false, (Bf), (short)0, (Cf), false, false)

// ---------------- GEMM: out[m][n] = sum_k X[m][k]*W[n][k] (+bias[n]) ----------------
__global__ __launch_bounds__(256)
void gemm_tn_wmma(const float* __restrict__ X, const float* __restrict__ W,
                  const float* __restrict__ bias, float* __restrict__ out,
                  int M, int K, int N) {
    const int lane = threadIdx.x & 31;
    const int wave = threadIdx.x >> 5;
    const int half = lane >> 4;
    const int lm   = lane & 15;
    const int m0 = (blockIdx.x * 8 + wave) * 16;
    const int n0 = blockIdx.y * 64;
    if (m0 >= M) return;

    v8f acc0 = {}, acc1 = {}, acc2 = {}, acc3 = {};
    const float* xrow = X + (size_t)(m0 + lm) * K;
    const float* wr0 = W + (size_t)(n0 +      lm) * K;
    const float* wr1 = W + (size_t)(n0 + 16 + lm) * K;
    const float* wr2 = W + (size_t)(n0 + 32 + lm) * K;
    const float* wr3 = W + (size_t)(n0 + 48 + lm) * K;

    for (int k0 = 0; k0 < K; k0 += 32) {
        v16bf a, b0, b1, b2, b3;
#pragma unroll
        for (int v = 0; v < 8; ++v) {
            int kb = k0 + kb_of(v, half);
            float2 xa = *(const float2*)(xrow + kb);
            float2 w0 = *(const float2*)(wr0 + kb);
            float2 w1 = *(const float2*)(wr1 + kb);
            float2 w2 = *(const float2*)(wr2 + kb);
            float2 w3 = *(const float2*)(wr3 + kb);
            a [2*v] = (__bf16)xa.x; a [2*v+1] = (__bf16)xa.y;
            b0[2*v] = (__bf16)w0.x; b0[2*v+1] = (__bf16)w0.y;
            b1[2*v] = (__bf16)w1.x; b1[2*v+1] = (__bf16)w1.y;
            b2[2*v] = (__bf16)w2.x; b2[2*v+1] = (__bf16)w2.y;
            b3[2*v] = (__bf16)w3.x; b3[2*v+1] = (__bf16)w3.y;
        }
        acc0 = WMMA_BF16(a, b0, acc0);
        acc1 = WMMA_BF16(a, b1, acc1);
        acc2 = WMMA_BF16(a, b2, acc2);
        acc3 = WMMA_BF16(a, b3, acc3);
    }
    float bv0 = bias ? bias[n0 +      lm] : 0.0f;
    float bv1 = bias ? bias[n0 + 16 + lm] : 0.0f;
    float bv2 = bias ? bias[n0 + 32 + lm] : 0.0f;
    float bv3 = bias ? bias[n0 + 48 + lm] : 0.0f;
#pragma unroll
    for (int r = 0; r < 8; ++r) {
        float* orow = out + (size_t)(m0 + r + 8 * half) * N + n0 + lm;
        orow[0]  = acc0[r] + bv0;
        orow[16] = acc1[r] + bv1;
        orow[32] = acc2[r] + bv2;
        orow[48] = acc3[r] + bv3;
    }
}

// ---------------- AdaIN stats ----------------
__global__ __launch_bounds__(320)
void adain_stats(const float* __restrict__ qkv, float* __restrict__ meanb,
                 float* __restrict__ stdb) {
    const int mb = blockIdx.x;
    const int c  = threadIdx.x;
    const float* base = qkv + (size_t)mb * SS * CC;
    float sum = 0.f, sq = 0.f;
    for (int s = 0; s < SS; ++s) {
        float x = base[(size_t)s * CC + c];
        sum += x; sq += x * x;
    }
    float mean = sum / (float)SS;
    float var  = (sq - sum * mean) / (float)(SS - 1);
    if (var < 0.f) var = 0.f;
    meanb[mb * CC + c] = mean;
    stdb [mb * CC + c] = sqrtf(var + 1e-5f);
}

// ---------------- AdaIN apply (in place, skip content rows 0 and 3) ----------------
__global__ __launch_bounds__(256)
void adain_apply(float* __restrict__ qkv, const float* __restrict__ meanb,
                 const float* __restrict__ stdb) {
    size_t idx = (size_t)blockIdx.x * 256 + threadIdx.x;
    int c = (int)(idx % CC);
    size_t sc = idx / CC;
    int bs = (int)(sc / SS);
    int b  = bs % BB;
    if (b == 0 || b == 3) return;
    int mat = bs / BB;
    int sb  = (b < 3) ? 1 : 4;
    int mb  = mat * BB + b, mbs = mat * BB + sb;
    float x = qkv[idx];
    qkv[idx] = (x - meanb[mb*CC + c]) / stdb[mb*CC + c] * stdb[mbs*CC + c]
               + meanb[mbs*CC + c];
}

// ---------------- f32 -> bf16 pre-convert (k and v, post-AdaIN) ----------------
__global__ __launch_bounds__(256)
void to_bf16_v4(const float* __restrict__ src, __bf16* __restrict__ dst) {
    size_t i = ((size_t)blockIdx.x * 256 + threadIdx.x) * 4;
    float4 f = *(const float4*)(src + i);
    bf4v o = { (__bf16)f.x, (__bf16)f.y, (__bf16)f.z, (__bf16)f.w };
    *(bf4v*)(dst + i) = o;
}

// ---------------- Flash attention ----------------
// Block = 8 waves sharing (b,h); each wave owns TWO 16-row Q tiles.
// K/V tiles are bf16 in global (pre-converted) and DMA'd into double-buffered
// LDS with global_load_async_to_lds_b128 (ASYNCcnt), overlapping the WMMA work.
__global__ __launch_bounds__(256)
void flash_attn(const float* __restrict__ qw, const __bf16* __restrict__ kb16,
                const __bf16* __restrict__ vb16, float* __restrict__ ow) {
    __shared__ __align__(16) __bf16 KtBuf[2][32 * KROW];   // d 40..71 pre-zeroed
    __shared__ __align__(16) __bf16 VtBuf[2][32 * VROW];   // d 40..47 pre-zeroed
    __shared__ __align__(16) __bf16 Pt[16 * 512];          // per (wave,qi) P staging

    const int tid  = threadIdx.x;
    const int lane = tid & 31, wave = tid >> 5;
    const int half = lane >> 4, lm = lane & 15;
    const int h = blockIdx.y, b = blockIdx.z;
    const int qt0 = (blockIdx.x * 8 + wave) * 2;
    const size_t bbase = (size_t)b * SS * CC + h * DH;     // element offset (f32 or bf16)
    const float scale = 0.15811388300841898f;              // 1/sqrt(40)

    // Issue one K/V tile fill via async b128 copies. 320 chunks of 16B:
    // c in [0,160): K row c/5, 8-elem seg c%5 ; c in [160,320): same for V.
    auto issue_tile = [&](int bufi, int kb0) {
#pragma unroll
        for (int ci = 0; ci < 2; ++ci) {
            int c = tid + ci * 256;
            if (c < 320) {
                int isV = (c >= 160);
                int cc  = isV ? c - 160 : c;
                int row = cc / 5, d0 = (cc % 5) * 8;
                unsigned long long g;
                unsigned lds;
                if (!isV) {
                    g   = (unsigned long long)(size_t)(kb16 + bbase + (size_t)(kb0 + row) * CC + d0);
                    lds = (unsigned)(size_t)&KtBuf[bufi][row * KROW + d0];
                } else {
                    g   = (unsigned long long)(size_t)(vb16 + bbase + (size_t)(kb0 + row) * CC + d0);
                    lds = (unsigned)(size_t)&VtBuf[bufi][row * VROW + d0];
                }
                asm volatile("global_load_async_to_lds_b128 %0, %1, off"
                             :: "v"(lds), "v"(g) : "memory");
            }
        }
    };

    // Q tiles in A-layout (pre-scaled), D padded 40 -> 64
    v16bf a_lo[2], a_hi[2];
#pragma unroll
    for (int qi = 0; qi < 2; ++qi) {
        const float* qrow = qw + bbase + (size_t)((qt0 + qi) * 16 + lm) * CC;
#pragma unroll
        for (int v = 0; v < 8; ++v) {
            int kb = kb_of(v, half);
            float2 xq = *(const float2*)(qrow + kb);
            a_lo[qi][2*v]   = (__bf16)(xq.x * scale);
            a_lo[qi][2*v+1] = (__bf16)(xq.y * scale);
            int kh = kb + 32;
            float y0 = (kh     < DH) ? qrow[kh]     * scale : 0.f;
            float y1 = (kh + 1 < DH) ? qrow[kh + 1] * scale : 0.f;
            a_hi[qi][2*v]   = (__bf16)y0;
            a_hi[qi][2*v+1] = (__bf16)y1;
        }
    }

    // zero pad columns once (async copies never touch them)
    for (int i = tid; i < 2 * 32 * 32; i += 256) {        // Kt: d 40..71, both buffers
        int bufi = i >> 10, r = (i >> 5) & 31, d = 40 + (i & 31);
        KtBuf[bufi][r * KROW + d] = (__bf16)0.f;
    }
    for (int i = tid; i < 2 * 32 * 8; i += 256) {         // Vt: d 40..47, both buffers
        int bufi = i >> 8, r = (i >> 3) & 31, d = 40 + (i & 7);
        VtBuf[bufi][r * VROW + d] = (__bf16)0.f;
    }

    float mrow[2][8], lrow[2][8];
    v8f oacc[2][3];
#pragma unroll
    for (int qi = 0; qi < 2; ++qi) {
#pragma unroll
        for (int r = 0; r < 8; ++r) { mrow[qi][r] = -3.0e38f; lrow[qi][r] = 0.f; }
#pragma unroll
        for (int t = 0; t < 3; ++t) oacc[qi][t] = (v8f){};
    }

    // prologue: fill buffer 0
    issue_tile(0, 0);
    asm volatile("s_wait_asynccnt 0" ::: "memory");
    __syncthreads();

    for (int kb0 = 0; kb0 < SS; kb0 += 32) {
        const int cur = (kb0 >> 5) & 1;
        if (kb0 + 32 < SS) issue_tile(cur ^ 1, kb0 + 32);  // overlap next fill with compute
        const __bf16* Kt = KtBuf[cur];
        const __bf16* Vt = VtBuf[cur];

        // ---- QK^T B fragments (shared by both q tiles) ----
        v16bf blo[2], bhi[2];
#pragma unroll
        for (int t = 0; t < 2; ++t) {
            int key = t * 16 + lm;
#pragma unroll
            for (int v = 0; v < 8; ++v) {
                int kb = kb_of(v, half);
                bf2 plo = *(const bf2*)&Kt[key * KROW + kb];
                bf2 phi = *(const bf2*)&Kt[key * KROW + kb + 32];
                blo[t][2*v] = plo[0]; blo[t][2*v+1] = plo[1];
                bhi[t][2*v] = phi[0]; bhi[t][2*v+1] = phi[1];
            }
        }
        // ---- PV B fragments (shared by both q tiles) ----
        v16bf bv[3];
#pragma unroll
        for (int t = 0; t < 3; ++t) {
            int d = t * 16 + lm;
#pragma unroll
            for (int v = 0; v < 8; ++v) {
                int kb = kb_of(v, half);
                bv[t][2*v]   = Vt[kb * VROW + d];
                bv[t][2*v+1] = Vt[(kb + 1) * VROW + d];
            }
        }

#pragma unroll
        for (int qi = 0; qi < 2; ++qi) {
            v8f s0 = {}, s1 = {};
            s0 = WMMA_BF16(a_lo[qi], blo[0], s0);
            s0 = WMMA_BF16(a_hi[qi], bhi[0], s0);
            s1 = WMMA_BF16(a_lo[qi], blo[1], s1);
            s1 = WMMA_BF16(a_hi[qi], bhi[1], s1);

            // online softmax: row m = r + 8*half lives in a 16-lane half
#pragma unroll
            for (int r = 0; r < 8; ++r) {
                float tm = fmaxf(s0[r], s1[r]);
                tm = fmaxf(tm, __shfl_xor(tm, 1, 32));
                tm = fmaxf(tm, __shfl_xor(tm, 2, 32));
                tm = fmaxf(tm, __shfl_xor(tm, 4, 32));
                tm = fmaxf(tm, __shfl_xor(tm, 8, 32));
                float nm = fmaxf(mrow[qi][r], tm);
                float alpha = __expf(mrow[qi][r] - nm);
                mrow[qi][r] = nm;
                float p0 = __expf(s0[r] - nm);
                float p1 = __expf(s1[r] - nm);
                s0[r] = p0; s1[r] = p1;
                float ps = p0 + p1;
                ps += __shfl_xor(ps, 1, 32);
                ps += __shfl_xor(ps, 2, 32);
                ps += __shfl_xor(ps, 4, 32);
                ps += __shfl_xor(ps, 8, 32);
                lrow[qi][r] = lrow[qi][r] * alpha + ps;
                oacc[qi][0][r] *= alpha;
                oacc[qi][1][r] *= alpha;
                oacc[qi][2][r] *= alpha;
            }

            // P: C-layout -> per-(wave,qi) LDS -> A-layout bf16 (intra-wave DS in order)
            __bf16* pw = Pt + (wave * 2 + qi) * 512;
#pragma unroll
            for (int r = 0; r < 8; ++r) {
                int m = r + 8 * half;
                pw[m * 32 + lm]      = (__bf16)s0[r];
                pw[m * 32 + 16 + lm] = (__bf16)s1[r];
            }
            v16bf pa;
#pragma unroll
            for (int v = 0; v < 8; ++v) {
                int kb = kb_of(v, half);
                bf2 pp = *(const bf2*)&pw[lm * 32 + kb];
                pa[2*v] = pp[0]; pa[2*v+1] = pp[1];
            }

            oacc[qi][0] = WMMA_BF16(pa, bv[0], oacc[qi][0]);
            oacc[qi][1] = WMMA_BF16(pa, bv[1], oacc[qi][1]);
            oacc[qi][2] = WMMA_BF16(pa, bv[2], oacc[qi][2]);
        }

        // drain this wave's async issues, then publish to all waves
        asm volatile("s_wait_asynccnt 0" ::: "memory");
        __syncthreads();
    }

    // epilogue: normalize and scatter into [B,S,C] with head offset
#pragma unroll
    for (int qi = 0; qi < 2; ++qi) {
#pragma unroll
        for (int r = 0; r < 8; ++r) {
            float inv = 1.0f / lrow[qi][r];
            float* orow = ow + bbase + (size_t)((qt0 + qi) * 16 + r + 8 * half) * CC;
            orow[lm]      = oacc[qi][0][r] * inv;
            orow[16 + lm] = oacc[qi][1][r] * inv;
            if (32 + lm < DH) orow[32 + lm] = oacc[qi][2][r] * inv;
        }
    }
}

extern "C" void kernel_launch(void* const* d_in, const int* in_sizes, int n_in,
                              void* d_out, int out_size, void* d_ws, size_t ws_size,
                              hipStream_t stream) {
    const float* hs = (const float*)d_in[0];
    const float* Wq = (const float*)d_in[1];
    const float* Wk = (const float*)d_in[2];
    const float* Wv = (const float*)d_in[3];
    const float* Wo = (const float*)d_in[4];
    const float* bo = (const float*)d_in[5];
    float* out = (float*)d_out;

    const size_t N1 = (size_t)BB * SS * CC;     // 3,932,160
    float* qw = (float*)d_ws;                   // q/k/v contiguous (stats + bf16 convert rely on it)
    float* kw = qw + N1;
    float* vw = kw + N1;
    float* ow = vw + N1;
    float* meanb = ow + N1;                     // 3*B*C
    float* stdb  = meanb + 3 * BB * CC;
    __bf16* kb16 = (__bf16*)(stdb + 3 * BB * CC);   // 16B-aligned by construction
    __bf16* vb16 = kb16 + N1;

    dim3 gg(96, 5, 1);
    gemm_tn_wmma<<<gg, 256, 0, stream>>>(hs, Wq, nullptr, qw, BB * SS, CC, CC);
    gemm_tn_wmma<<<gg, 256, 0, stream>>>(hs, Wk, nullptr, kw, BB * SS, CC, CC);
    gemm_tn_wmma<<<gg, 256, 0, stream>>>(hs, Wv, nullptr, vw, BB * SS, CC, CC);

    adain_stats<<<3 * BB, 320, 0, stream>>>(qw, meanb, stdb);
    adain_apply<<<(unsigned)((3 * N1) / 256), 256, 0, stream>>>(qw, meanb, stdb);

    // k and v are contiguous: convert both with one launch (2*N1 elems, 4/thread)
    to_bf16_v4<<<(unsigned)((2 * N1) / (256 * 4)), 256, 0, stream>>>(kw, kb16);

    flash_attn<<<dim3(8, HH, BB), 256, 0, stream>>>(qw, kb16, vb16, ow);

    gemm_tn_wmma<<<gg, 256, 0, stream>>>(ow, Wo, bo, out, BB * SS, CC, CC);
}
